// LSTM_Base_81492709474431
// MI455X (gfx1250) — matrix operand order repaired and checked by reading, not compiled, not run on hardware
//
#include <hip/hip_runtime.h>
#include <hip/hip_bf16.h>
#include <math.h>

// ---------------------------------------------------------------------------
// LSTM forward for MI455X (gfx1250), bf16 WMMA with f32 accumulation.
// T=512, B=64, I=512, H=1024, gates [f,i,g,o].
// ---------------------------------------------------------------------------

typedef __bf16 bf16_t;
typedef __attribute__((ext_vector_type(16))) __bf16 v16bf;
typedef __attribute__((ext_vector_type(8)))  __bf16 v8bf;
typedef __attribute__((ext_vector_type(8)))  float  v8f;

#define T_STEPS 512
#define BATCH   64
#define IDIM    512
#define HDIM    1024

// ---------------------------------------------------------------------------
// Prep kernels (one-time per call; HBM-bound, negligible vs recurrent loop)
// ---------------------------------------------------------------------------

__global__ void cvt_bf16_kernel(const float* __restrict__ in,
                                bf16_t* __restrict__ out, int n) {
  int i = blockIdx.x * blockDim.x + threadIdx.x;
  if (i < n) out[i] = (bf16_t)in[i];
}

// in: [G][R][C] f32  ->  out: [G][C][R] bf16   (weights become K-contiguous)
__global__ void cvt_transpose_kernel(const float* __restrict__ in,
                                     bf16_t* __restrict__ out,
                                     int R, int C, int total) {
  int i = blockIdx.x * blockDim.x + threadIdx.x;
  if (i >= total) return;
  int g   = i / (R * C);
  int rem = i - g * (R * C);
  int rr  = rem / C;
  int cc  = rem - rr * C;
  out[(size_t)g * R * C + (size_t)cc * R + rr] = (bf16_t)in[i];
}

__global__ void init_state_kernel(const float* __restrict__ h0,
                                  const float* __restrict__ c0,
                                  bf16_t* __restrict__ hbf,
                                  float* __restrict__ cbuf, int n) {
  int i = blockIdx.x * blockDim.x + threadIdx.x;
  if (i < n) { hbf[i] = (bf16_t)h0[i]; cbuf[i] = c0[i]; }
}

__global__ void bias_kernel(const float* __restrict__ bi,
                            const float* __restrict__ bh,
                            float* __restrict__ bias, int n) {
  int i = blockIdx.x * blockDim.x + threadIdx.x;
  if (i < n) bias[i] = bi[i] + bh[i];
}

// ---------------------------------------------------------------------------
// Recurrent step: grid = H/16 = 64 blocks, 128 threads = 4 waves (wave = gate).
// Each wave computes z_g for all 64 batch rows x 16 H-columns via WMMA,
// K = I (x-part) + H (h-part). Gate fusion via LDS, cell update, bf16 h out.
// ---------------------------------------------------------------------------

__device__ __forceinline__ float sigmoidf_fast(float x) {
  return 1.0f / (1.0f + __expf(-x));
}

__global__ __launch_bounds__(128)
void lstm_step_kernel(int t,
                      const bf16_t* __restrict__ Xbf,   // [T][B][I]
                      const bf16_t* __restrict__ WiT,   // [4][H][I]
                      const bf16_t* __restrict__ WhT,   // [4][H][H]
                      const float*  __restrict__ bias,  // [4][H]
                      const bf16_t* __restrict__ hprev, // [B][H] bf16
                      bf16_t*       __restrict__ hnext, // [B][H] bf16
                      float*        __restrict__ cbuf,  // [B][H] f32 (in/out)
                      float*        __restrict__ out)   // [T][B][H] f32
{
  const int g     = threadIdx.x >> 5;   // gate index 0..3 (f,i,g,o)
  const int lane  = threadIdx.x & 31;
  const int r     = lane & 15;          // row-in-tile / column-in-tile
  const int hi    = lane >> 4;          // lane half
  const int nBase = blockIdx.x * 16;    // H-column tile

  __shared__ float zsh[4][BATCH][16];   // z per gate for the whole batch tile

  // C/D layout: VGPR v -> M = v + 8*hi, N = r. Bias depends only on N.
  const float bb = bias[g * HDIM + nBase + r];
  v8f acc[4];
  #pragma unroll
  for (int mt = 0; mt < 4; ++mt) {
    #pragma unroll
    for (int e = 0; e < 8; ++e) acc[mt][e] = bb;
  }

  // ---------------- input projection: K over IDIM --------------------------
  {
    const bf16_t* __restrict__ xr = Xbf + (size_t)t * BATCH * IDIM;
    const bf16_t* __restrict__ wr =
        WiT + ((size_t)g * HDIM + (size_t)(nBase + r)) * IDIM;
    #pragma unroll 2
    for (int kk = 0; kk < IDIM; kk += 32) {
      // B fragment (32x16): lane holds column N=r, K = {hi*8+0..7, 16+hi*8+0..7}
      const v8bf b0 = *(const v8bf*)(wr + kk + hi * 8);
      const v8bf b1 = *(const v8bf*)(wr + kk + 16 + hi * 8);
      const v16bf bfrag = __builtin_shufflevector(
          b0, b1, 0,1,2,3,4,5,6,7,8,9,10,11,12,13,14,15);
      #pragma unroll
      for (int mt = 0; mt < 4; ++mt) {
        // A fragment (16x32): row M = mt*16 + r, K = {hi*8+0..7, 16+hi*8+0..7}
        const bf16_t* __restrict__ xa =
            xr + (size_t)(mt * 16 + r) * IDIM + kk;
        const v8bf a0 = *(const v8bf*)(xa + hi * 8);
        const v8bf a1 = *(const v8bf*)(xa + 16 + hi * 8);
        const v16bf afrag = __builtin_shufflevector(
            a0, a1, 0,1,2,3,4,5,6,7,8,9,10,11,12,13,14,15);
        acc[mt] = __builtin_amdgcn_wmma_f32_16x16x32_bf16(
            false, afrag, false, bfrag, (short)0, acc[mt], false, false);
      }
    }
  }

  // ---------------- hidden projection: K over HDIM --------------------------
  {
    const bf16_t* __restrict__ wr =
        WhT + ((size_t)g * HDIM + (size_t)(nBase + r)) * HDIM;
    #pragma unroll 2
    for (int kk = 0; kk < HDIM; kk += 32) {
      const v8bf b0 = *(const v8bf*)(wr + kk + hi * 8);
      const v8bf b1 = *(const v8bf*)(wr + kk + 16 + hi * 8);
      const v16bf bfrag = __builtin_shufflevector(
          b0, b1, 0,1,2,3,4,5,6,7,8,9,10,11,12,13,14,15);
      #pragma unroll
      for (int mt = 0; mt < 4; ++mt) {
        const bf16_t* __restrict__ ha =
            hprev + (size_t)(mt * 16 + r) * HDIM + kk;
        const v8bf a0 = *(const v8bf*)(ha + hi * 8);
        const v8bf a1 = *(const v8bf*)(ha + 16 + hi * 8);
        const v16bf afrag = __builtin_shufflevector(
            a0, a1, 0,1,2,3,4,5,6,7,8,9,10,11,12,13,14,15);
        acc[mt] = __builtin_amdgcn_wmma_f32_16x16x32_bf16(
            false, afrag, false, bfrag, (short)0, acc[mt], false, false);
      }
    }
  }

  // ---------------- gate fusion via LDS -------------------------------------
  #pragma unroll
  for (int mt = 0; mt < 4; ++mt) {
    #pragma unroll
    for (int v = 0; v < 8; ++v) {
      zsh[g][mt * 16 + v + 8 * hi][r] = acc[mt][v];
    }
  }
  __syncthreads();

  // ---------------- cell update: 1024 elements over 128 threads -------------
  for (int e = threadIdx.x; e < BATCH * 16; e += 128) {
    const int b = e >> 4;
    const int n = e & 15;
    const float zf = zsh[0][b][n];
    const float zi = zsh[1][b][n];
    const float zg = zsh[2][b][n];
    const float zo = zsh[3][b][n];
    const float fg = sigmoidf_fast(zf);
    const float ig = sigmoidf_fast(zi);
    const float gg = tanhf(zg);
    const float og = sigmoidf_fast(zo);
    const size_t idx = (size_t)b * HDIM + (size_t)(nBase + n);
    const float cn = fg * cbuf[idx] + ig * gg;
    const float hn = og * tanhf(cn);
    cbuf[idx] = cn;
    out[(size_t)t * BATCH * HDIM + idx] = hn;
    hnext[idx] = (bf16_t)hn;
  }
}

// ---------------------------------------------------------------------------
// Tail: h_n = outputs[T-1], c_n = cbuf
// ---------------------------------------------------------------------------

__global__ void tail_kernel(const float* __restrict__ cbuf,
                            float* __restrict__ out, int n) {
  int i = blockIdx.x * blockDim.x + threadIdx.x;
  if (i < n) {
    const size_t TBH = (size_t)T_STEPS * BATCH * HDIM;
    out[TBH + i]     = out[(size_t)(T_STEPS - 1) * BATCH * HDIM + i]; // h_n
    out[TBH + n + i] = cbuf[i];                                       // c_n
  }
}

// ---------------------------------------------------------------------------
// Launcher
// ---------------------------------------------------------------------------

extern "C" void kernel_launch(void* const* d_in, const int* in_sizes, int n_in,
                              void* d_out, int out_size, void* d_ws, size_t ws_size,
                              hipStream_t stream) {
  (void)in_sizes; (void)n_in; (void)out_size; (void)ws_size;

  const float* X  = (const float*)d_in[0];  // [T,B,I]
  const float* h0 = (const float*)d_in[1];  // [B,H]
  const float* c0 = (const float*)d_in[2];  // [B,H]
  const float* Wi = (const float*)d_in[3];  // [4,I,H]
  const float* Wh = (const float*)d_in[4];  // [4,H,H]
  const float* bi = (const float*)d_in[5];  // [4,H]
  const float* bh = (const float*)d_in[6];  // [4,H]
  float* out = (float*)d_out;               // [T,B,H] ++ [B,H] ++ [B,H]

  // Workspace layout (bytes, 256-aligned blocks):
  char* ws = (char*)d_ws;
  bf16_t* Xbf  = (bf16_t*)(ws);                 // T*B*I*2   = 33,554,432
  bf16_t* WiT  = (bf16_t*)(ws + 33554432);      // 4*H*I*2   =  4,194,304
  bf16_t* WhT  = (bf16_t*)(ws + 37748736);      // 4*H*H*2   =  8,388,608
  bf16_t* hA   = (bf16_t*)(ws + 46137344);      // B*H*2     =    131,072
  bf16_t* hB   = (bf16_t*)(ws + 46268416);      // B*H*2     =    131,072
  float*  cbuf = (float*) (ws + 46399488);      // B*H*4     =    262,144
  float*  bias = (float*) (ws + 46661632);      // 4*H*4     =     16,384
                                                // total ~44.6 MB

  const int nX  = T_STEPS * BATCH * IDIM;       // 16,777,216
  const int nWi = 4 * IDIM * HDIM;              //  2,097,152
  const int nWh = 4 * HDIM * HDIM;              //  4,194,304
  const int nBH = BATCH * HDIM;                 //     65,536
  const int nG  = 4 * HDIM;                     //      4,096

  cvt_bf16_kernel<<<nX / 256, 256, 0, stream>>>(X, Xbf, nX);
  cvt_transpose_kernel<<<nWi / 256, 256, 0, stream>>>(Wi, WiT, IDIM, HDIM, nWi);
  cvt_transpose_kernel<<<nWh / 256, 256, 0, stream>>>(Wh, WhT, HDIM, HDIM, nWh);
  init_state_kernel<<<nBH / 256, 256, 0, stream>>>(h0, c0, hA, cbuf, nBH);
  bias_kernel<<<nG / 256, 256, 0, stream>>>(bi, bh, bias, nG);

  for (int t = 0; t < T_STEPS; ++t) {
    const bf16_t* hp = (t & 1) ? hB : hA;
    bf16_t*       hn = (t & 1) ? hA : hB;
    lstm_step_kernel<<<HDIM / 16, 128, 0, stream>>>(
        t, Xbf, WiT, WhT, bias, hp, hn, cbuf, out);
  }

  tail_kernel<<<nBH / 256, 256, 0, stream>>>(cbuf, out, nBH);
}
